// Net2_38963943309314
// MI455X (gfx1250) — matrix operand order repaired and checked by reading, not compiled, not run on hardware
//
#include <hip/hip_runtime.h>
#include <hip/hip_bf16.h>
#include <stdint.h>

// ---------------------------------------------------------------------------
// Sparse submanifold CNN trunk for MI455X (gfx1250, wave32, WMMA).
// hash grid -> rules(27)+tile masks -> conv(32pad->256) -> conv(256->256)
//   -> sorted-unique 2x2x2 maxpool -> rules(27)@half-res
//   -> conv(256->512) -> conv(512->512) -> d_out (f32) + pcoors (i32).
// Weights pre-repacked into per-lane WMMA fragment order (B = 4x global b128
// loads/lane); A staged in LDS in fragment order (2x ds_load_b128/lane).
// Per-16-row-tile 27-bit offset masks make inactive offsets cost zero.
// ---------------------------------------------------------------------------

typedef __attribute__((ext_vector_type(16))) __bf16          v16bf;
typedef __attribute__((ext_vector_type(16))) unsigned short  v16u;
typedef __attribute__((ext_vector_type(8)))  float           v8f;

#define DGRID 40
#define HGRID 1024
#define WGRID 1024
#define D2G   20
#define H2G   512
#define W2G   512
#define P2TOT (D2G * H2G * W2G)      // 5,242,880 parent sites
#define NBLK  (P2TOT / 1024)         // 5120 scan blocks (exact)
#define HSIZE (1 << 17)              // hash entries (load factor ~0.38)

// ---- bf16 helpers (RNE) ----------------------------------------------------
static __device__ __forceinline__ unsigned short f2bf(float f) {
  union { float f; unsigned u; } v; v.f = f;
  unsigned r = v.u + 0x7FFFu + ((v.u >> 16) & 1u);
  return (unsigned short)(r >> 16);
}
static __device__ __forceinline__ float bf2f(unsigned short h) {
  union { unsigned u; float f; } v; v.u = ((unsigned)h) << 16;
  return v.f;
}

// ---- hash table ------------------------------------------------------------
static __device__ __forceinline__ unsigned hmix(int key) {
  unsigned h = (unsigned)key * 0x9E3779B1u;
  h ^= h >> 15;
  return h & (HSIZE - 1);
}
static __device__ __forceinline__ void hinsert(int* keys, int* vals, int key, int val) {
  unsigned h = hmix(key);
  for (;;) {
    int old = atomicCAS(&keys[h], -1, key);
    if (old == -1 || old == key) { vals[h] = val; return; }
    h = (h + 1) & (HSIZE - 1);
  }
}
static __device__ __forceinline__ int hlookup(const int* keys, const int* vals, int key) {
  unsigned h = hmix(key);
  for (int it = 0; it < HSIZE; ++it) {
    int k = keys[h];
    if (k == key) return vals[h];
    if (k == -1)  return -1;
    h = (h + 1) & (HSIZE - 1);
  }
  return -1;
}

// ---- small utility kernels -------------------------------------------------
__global__ void k_seti(int* p, int v, int n) {
  int i = blockIdx.x * blockDim.x + threadIdx.x;
  if (i < n) p[i] = v;
}

// Pad features [n][3] f32 -> [n][32] bf16 (zeros beyond ch 3).
__global__ void k_padfeat(const float* __restrict__ s, unsigned short* __restrict__ d, int n) {
  int gid = blockIdx.x * blockDim.x + threadIdx.x;
  if (gid >= n * 32) return;
  int i = gid >> 5, c = gid & 31;
  d[gid] = (c < 3) ? f2bf(s[i * 3 + c]) : (unsigned short)0;
}

// Repack weights [27][Cin][Cout] f32 -> WMMA B-fragment order bf16:
// dst[((k*(Cout/16)+ct)*KT + kt)*512 + lane*16 + e]
//   = w[k][kt*32 + (lane>=16 ? 16:0) + e][ct*16 + (lane&15)]   (0 if cin>=Cin)
__global__ void k_repack(const float* __restrict__ src, unsigned short* __restrict__ dst,
                         int Cin, int Cout, int KT, int total) {
  int gid = blockIdx.x * blockDim.x + threadIdx.x;
  if (gid >= total) return;
  int e    = gid & 15;
  int lane = (gid >> 4) & 31;
  int rest = gid >> 9;
  int kt   = rest % KT;
  int ct   = (rest / KT) % (Cout / 16);
  int k    = rest / (KT * (Cout / 16));
  int cin  = kt * 32 + ((lane >= 16) ? 16 : 0) + e;
  int cout = ct * 16 + (lane & 15);
  dst[gid] = (cin < Cin) ? f2bf(src[((size_t)k * Cin + cin) * Cout + cout])
                         : (unsigned short)0;
}

__global__ void k_hash_insert(const int* __restrict__ coors, int n, int* keys, int* vals) {
  int i = blockIdx.x * blockDim.x + threadIdx.x;
  if (i >= n) return;
  int key = (coors[i * 3 + 0] * HGRID + coors[i * 3 + 1]) * WGRID + coors[i * 3 + 2];
  hinsert(keys, vals, key, i);
}

// rules[k*n + i] = row of neighbor at OFF3[k], -1 if inactive.
// Also ORs bit k into tileMask[i/16] when a neighbor exists.
__global__ void k_rules(const int* __restrict__ coors, int n, const int* limitPtr,
                        int Dd, int Hh, int Ww,
                        const int* __restrict__ keys, const int* __restrict__ vals,
                        int* __restrict__ rules, unsigned* __restrict__ tileMask) {
  int gid = blockIdx.x * blockDim.x + threadIdx.x;
  if (gid >= n * 27) return;
  int i = gid % n, k = gid / n;
  int lim = limitPtr ? *limitPtr : n;
  int res = -1;
  if (i < lim) {
    int dz = k / 9 - 1, dy = (k / 3) % 3 - 1, dx = k % 3 - 1;
    int nz = coors[i * 3 + 0] + dz, ny = coors[i * 3 + 1] + dy, nx = coors[i * 3 + 2] + dx;
    if (nz >= 0 && nz < Dd && ny >= 0 && ny < Hh && nx >= 0 && nx < Ww)
      res = hlookup(keys, vals, (nz * Hh + ny) * Ww + nx);
  }
  rules[gid] = res;
  if (res >= 0) atomicOr(&tileMask[i >> 4], 1u << k);
}

// ---- pooling: flags -> two-level exclusive scan -> sorted compaction -------
__global__ void k_flags(const int* __restrict__ coors, int n, int* __restrict__ flags) {
  int i = blockIdx.x * blockDim.x + threadIdx.x;
  if (i >= n) return;
  int p = ((coors[i * 3 + 0] / 2) * H2G + coors[i * 3 + 1] / 2) * W2G + coors[i * 3 + 2] / 2;
  flags[p] = 1;
}

__global__ void k_scan_block(const int* __restrict__ flags, int* __restrict__ scan,
                             int* __restrict__ bsums) {
  __shared__ int sh[1024];
  int t = threadIdx.x;
  int g = blockIdx.x * 1024 + t;
  int v = flags[g];
  sh[t] = v;
  __syncthreads();
  for (int off = 1; off < 1024; off <<= 1) {
    int x = 0;
    if (t >= off) x = sh[t - off];
    __syncthreads();
    if (t >= off) sh[t] += x;
    __syncthreads();
  }
  scan[g] = sh[t] - v;
  if (t == 1023) bsums[blockIdx.x] = sh[t];
}

__global__ void k_scan_top(int* bsums, int nb, int* mOut) {
  __shared__ int sh[1024];
  __shared__ int carry;
  int t = threadIdx.x;
  if (t == 0) carry = 0;
  __syncthreads();
  for (int c = 0; c < nb; c += 1024) {
    int v = (c + t < nb) ? bsums[c + t] : 0;
    sh[t] = v;
    __syncthreads();
    for (int off = 1; off < 1024; off <<= 1) {
      int x = 0;
      if (t >= off) x = sh[t - off];
      __syncthreads();
      if (t >= off) sh[t] += x;
      __syncthreads();
    }
    int base = carry;
    if (c + t < nb) bsums[c + t] = sh[t] - v + base;
    int tot = sh[1023];
    __syncthreads();
    if (t == 0) carry = base + tot;
    __syncthreads();
  }
  if (t == 0) mOut[0] = carry;
}

__global__ void k_finalize(const int* __restrict__ flags, const int* __restrict__ scan,
                           const int* __restrict__ bofs,
                           int* keys2, int* vals2, int* __restrict__ pcoors) {
  int p = blockIdx.x * blockDim.x + threadIdx.x;
  if (p >= P2TOT || !flags[p]) return;
  int row = scan[p] + bofs[p >> 10];
  hinsert(keys2, vals2, p, row);
  int pz = p / (H2G * W2G), py = (p / W2G) % H2G, px = p % W2G;
  pcoors[row * 3 + 0] = pz;
  pcoors[row * 3 + 1] = py;
  pcoors[row * 3 + 2] = px;
}

// ---- sparse maxpool (256 ch, one block per output row) ---------------------
__global__ void k_maxpool(const unsigned short* __restrict__ act,
                          const int* __restrict__ pcoors, const int* __restrict__ mPtr,
                          const int* __restrict__ keys, const int* __restrict__ vals,
                          unsigned short* __restrict__ out) {
  __shared__ int crow[8];
  int i = blockIdx.x, t = threadIdx.x;
  int M = *mPtr;
  if (i >= M) { out[(size_t)i * 256 + t] = 0; return; }
  if (t < 8) {
    int cz = pcoors[i * 3 + 0] * 2 + (t >> 2);
    int cy = pcoors[i * 3 + 1] * 2 + ((t >> 1) & 1);
    int cx = pcoors[i * 3 + 2] * 2 + (t & 1);
    crow[t] = hlookup(keys, vals, (cz * HGRID + cy) * WGRID + cx);
  }
  __syncthreads();
  float m = -1e30f;
  for (int k = 0; k < 8; ++k) {
    int r = crow[k];
    if (r >= 0) m = fmaxf(m, bf2f(act[(size_t)r * 256 + t]));
  }
  out[(size_t)i * 256 + t] = f2bf(m);
}

// ---- gather-GEMM, bf16 WMMA, f32 accumulate --------------------------------
// Block = 128 threads (4 waves). Tile: 16 rows x 128 cols; each wave owns two
// 16x16 sub-tiles (shared A fragment, two B fragments). The per-tile 27-bit
// offset mask is iterated with ctz, so inactive offsets cost nothing.
template <int CIN, int COUT, bool OUT_F32>
__global__ void k_gemm(const unsigned short* __restrict__ A,     // [n][CIN] bf16
                       const int* __restrict__ rules,            // [27][n]
                       const unsigned* __restrict__ tileMask,    // [n/16] 27-bit masks
                       const unsigned short* __restrict__ Wf,    // repacked fragments
                       void* __restrict__ Out,                   // bf16 [n][COUT] or f32
                       int n) {
  constexpr int KT = CIN / 32;
  __shared__ __align__(16) unsigned short AFrag[32][16];  // fragment-order A tile
  __shared__ int rulesSh[16];

  const int t    = threadIdx.x;
  const int lane = t & 31;
  const int wave = t >> 5;
  const int r0   = blockIdx.x * 16;
  const int ct0  = blockIdx.y * 8 + wave * 2;   // two 16-col tiles per wave

  v8f acc0 = {}, acc1 = {};

  // A staging geometry: this thread loads 4 consecutive K of one gathered row
  // and stores them at the fragment-inverse (lane,e) position.
  const int sm  = t >> 3;            // row 0..15
  const int skg = (t & 7) * 4;       // K group 0,4,...,28
  const int slane  = sm + ((skg & 16) ? 16 : 0);
  const int sebase = skg & 15;

  unsigned mask = tileMask[blockIdx.x];   // uniform per block

  while (mask) {
    const int k = (int)__builtin_ctz(mask);
    mask &= mask - 1;

    if (t < 16)
      rulesSh[t] = (r0 + t < n) ? rules[(size_t)k * n + r0 + t] : -1;
    __syncthreads();

    const unsigned short* WkA = Wf + ((size_t)(k * (COUT / 16) + ct0 + 0) * KT) * 512;
    const unsigned short* WkB = Wf + ((size_t)(k * (COUT / 16) + ct0 + 1) * KT) * 512;

    for (int kt = 0; kt < KT; ++kt) {
      // stage A (gathered, fragment order): one 8B store per thread
      {
        int r = rulesSh[sm];
        uint2 val = {0u, 0u};
        if (r >= 0)
          val = *(const uint2*)(A + (size_t)r * CIN + kt * 32 + skg);
        *(uint2*)&AFrag[slane][sebase] = val;
      }
      __syncthreads();

      union { v16u v; uint4 q[2]; } au, b0, b1;
      au.q[0] = ((const uint4*)&AFrag[lane][0])[0];
      au.q[1] = ((const uint4*)&AFrag[lane][0])[1];
      {
        const uint4* pa = (const uint4*)(WkA + (size_t)kt * 512 + lane * 16);
        const uint4* pb = (const uint4*)(WkB + (size_t)kt * 512 + lane * 16);
        b0.q[0] = pa[0]; b0.q[1] = pa[1];
        b1.q[0] = pb[0]; b1.q[1] = pb[1];
        if (kt + 1 < KT)
          __builtin_prefetch(WkA + (size_t)(kt + 1) * 512 + lane * 16, 0, 1);
      }
      v16bf af  = __builtin_bit_cast(v16bf, au.v);
      v16bf bf0 = __builtin_bit_cast(v16bf, b0.v);
      v16bf bf1 = __builtin_bit_cast(v16bf, b1.v);
      acc0 = __builtin_amdgcn_wmma_f32_16x16x32_bf16(false, af, false, bf0,
                                                     (short)0, acc0, false, false);
      acc1 = __builtin_amdgcn_wmma_f32_16x16x32_bf16(false, af, false, bf1,
                                                     (short)0, acc1, false, false);
      __syncthreads();   // protects AFrag (and rulesSh) for next iteration
    }
  }

  // epilogue: C/D layout -> lanes0-15 rows0..7 col=lane; lanes16-31 rows8..15
  union { v8f v; float f[8]; } u0, u1;
  u0.v = acc0; u1.v = acc1;
  int col0  = (ct0 + 0) * 16 + (lane & 15);
  int col1  = (ct0 + 1) * 16 + (lane & 15);
  int mbase = (lane < 16) ? 0 : 8;
  if (OUT_F32) {
    float* O = (float*)Out;
    for (int e = 0; e < 8; ++e) {
      int row = r0 + mbase + e;
      if (row < n) {
        O[(size_t)row * COUT + col0] = u0.f[e];
        O[(size_t)row * COUT + col1] = u1.f[e];
      }
    }
  } else {
    unsigned short* O = (unsigned short*)Out;
    for (int e = 0; e < 8; ++e) {
      int row = r0 + mbase + e;
      if (row < n) {
        O[(size_t)row * COUT + col0] = f2bf(u0.f[e]);
        O[(size_t)row * COUT + col1] = f2bf(u1.f[e]);
      }
    }
  }
}

// ---------------------------------------------------------------------------
extern "C" void kernel_launch(void* const* d_in, const int* in_sizes, int n_in,
                              void* d_out, int out_size, void* d_ws, size_t ws_size,
                              hipStream_t stream) {
  const float* features = (const float*)d_in[0];
  const float* w0f = (const float*)d_in[1];
  const float* w1f = (const float*)d_in[2];
  const float* w2f = (const float*)d_in[3];
  const float* w3f = (const float*)d_in[4];
  const int*   coors = (const int*)d_in[5];

  const int N = in_sizes[0] / 3;
  const int NT = (N + 15) / 16;   // 16-row tiles

  // repacked fragment sizes: 27 * (Cout/16) * KT * 512 bf16
  const int fr0 = 27 * (256 / 16) * 1  * 512;   // Cin 3 -> padded 32
  const int fr1 = 27 * (256 / 16) * 8  * 512;
  const int fr2 = 27 * (512 / 16) * 8  * 512;
  const int fr3 = 27 * (512 / 16) * 16 * 512;

  // ---- workspace layout ----------------------------------------------------
  char* ws = (char*)d_ws;
  size_t off = 0;
  auto alloc = [&](size_t bytes) -> char* {
    off = (off + 255) & ~(size_t)255;
    char* p = ws + off;
    off += bytes;
    return p;
  };
  unsigned short* wb0  = (unsigned short*)alloc((size_t)fr0 * 2);
  unsigned short* wb1  = (unsigned short*)alloc((size_t)fr1 * 2);
  unsigned short* wb2  = (unsigned short*)alloc((size_t)fr2 * 2);
  unsigned short* wb3  = (unsigned short*)alloc((size_t)fr3 * 2);
  unsigned short* actF = (unsigned short*)alloc((size_t)N * 32 * 2);
  unsigned short* act1 = (unsigned short*)alloc((size_t)N * 256 * 2);
  unsigned short* act2 = (unsigned short*)alloc((size_t)N * 256 * 2);
  unsigned short* actp = (unsigned short*)alloc((size_t)N * 256 * 2);
  unsigned short* act3 = (unsigned short*)alloc((size_t)N * 512 * 2);
  int* h1k    = (int*)alloc((size_t)HSIZE * 4);
  int* h1v    = (int*)alloc((size_t)HSIZE * 4);
  int* h2k    = (int*)alloc((size_t)HSIZE * 4);
  int* h2v    = (int*)alloc((size_t)HSIZE * 4);
  int* rules1 = (int*)alloc((size_t)27 * N * 4);
  int* rules2 = (int*)alloc((size_t)27 * N * 4);
  unsigned* tmask1 = (unsigned*)alloc((size_t)NT * 4);
  unsigned* tmask2 = (unsigned*)alloc((size_t)NT * 4);
  int* flags  = (int*)alloc((size_t)P2TOT * 4);
  int* scan   = (int*)alloc((size_t)P2TOT * 4);
  int* bsums  = (int*)alloc((size_t)NBLK * 4);
  int* mcnt   = (int*)alloc(4);

  float* xout  = (float*)d_out;                  // [N][512] f32
  int*   pcoor = (int*)(xout + (size_t)N * 512); // [N][3] i32

  const int B = 256;
  auto g1 = [&](int n) { return dim3((unsigned)((n + B - 1) / B)); };

  // 1) weight repack + feature pad (bf16, fragment order)
  k_repack<<<g1(fr0), B, 0, stream>>>(w0f, wb0, 3,   256, 1,  fr0);
  k_repack<<<g1(fr1), B, 0, stream>>>(w1f, wb1, 256, 256, 8,  fr1);
  k_repack<<<g1(fr2), B, 0, stream>>>(w2f, wb2, 256, 512, 8,  fr2);
  k_repack<<<g1(fr3), B, 0, stream>>>(w3f, wb3, 512, 512, 16, fr3);
  k_padfeat<<<g1(N * 32), B, 0, stream>>>(features, actF, N);

  // 2) level-1 hash + rules (+ tile masks)
  k_seti<<<g1(HSIZE), B, 0, stream>>>(h1k, -1, HSIZE);
  k_seti<<<g1(NT), B, 0, stream>>>((int*)tmask1, 0, NT);
  k_hash_insert<<<g1(N), B, 0, stream>>>(coors, N, h1k, h1v);
  k_rules<<<g1(N * 27), B, 0, stream>>>(coors, N, nullptr, DGRID, HGRID, WGRID,
                                        h1k, h1v, rules1, tmask1);

  // 3) conv 32pad->256, conv 256->256 (WMMA)
  dim3 gg1((N + 15) / 16, 256 / 128);
  k_gemm<32,  256, false><<<gg1, 128, 0, stream>>>(actF, rules1, tmask1, wb0, act1, N);
  k_gemm<256, 256, false><<<gg1, 128, 0, stream>>>(act1, rules1, tmask1, wb1, act2, N);

  // 4) sorted-unique pooling (flags -> scan -> compaction), hash2, pcoors
  k_seti<<<g1(P2TOT), B, 0, stream>>>(flags, 0, P2TOT);
  k_seti<<<g1(HSIZE), B, 0, stream>>>(h2k, -1, HSIZE);
  k_seti<<<g1(N * 3), B, 0, stream>>>(pcoor, 0, N * 3);
  k_flags<<<g1(N), B, 0, stream>>>(coors, N, flags);
  k_scan_block<<<NBLK, 1024, 0, stream>>>(flags, scan, bsums);
  k_scan_top<<<1, 1024, 0, stream>>>(bsums, NBLK, mcnt);
  k_finalize<<<g1(P2TOT), B, 0, stream>>>(flags, scan, bsums, h2k, h2v, pcoor);

  // 5) maxpool gather (children via level-1 hash)
  k_maxpool<<<N, 256, 0, stream>>>(act2, pcoor, mcnt, h1k, h1v, actp);

  // 6) level-2 rules (+ masks), conv 256->512, conv 512->512 (final f32)
  k_seti<<<g1(NT), B, 0, stream>>>((int*)tmask2, 0, NT);
  k_rules<<<g1(N * 27), B, 0, stream>>>(pcoor, N, mcnt, D2G, H2G, W2G,
                                        h2k, h2v, rules2, tmask2);
  dim3 gg2((N + 15) / 16, 512 / 128);
  k_gemm<256, 512, false><<<gg2, 128, 0, stream>>>(actp, rules2, tmask2, wb2, act3, N);
  k_gemm<512, 512, true ><<<gg2, 128, 0, stream>>>(act3, rules2, tmask2, wb3, (void*)xout, N);
}
